// CategoricalGATPolicy_12936441496281
// MI455X (gfx1250) — matrix operand order repaired and checked by reading, not compiled, not run on hardware
//
#include <hip/hip_runtime.h>

// ---------------------------------------------------------------------------
// Types for CDNA5 WMMA (wave32): v_wmma_f32_16x16x32_bf16
// ---------------------------------------------------------------------------
typedef __attribute__((ext_vector_type(16))) __bf16 v16bf;
typedef __attribute__((ext_vector_type(8)))  __bf16 v8bf;
typedef __attribute__((ext_vector_type(8)))  float  v8f;

#define F_IN 128
#define HIDC 64
#define HEADS 4
#define ACT 32
#define SENT 768
#define NB 4   // N-tiles per wave in the WMMA GEMM (A-fragment reuse factor)

static __device__ __forceinline__ unsigned f2ord(float f) {
    unsigned u = __float_as_uint(f);
    return (u & 0x80000000u) ? ~u : (u | 0x80000000u);
}
static __device__ __forceinline__ float ord2f(unsigned u) {
    unsigned v = (u & 0x80000000u) ? (u & 0x7FFFFFFFu) : ~u;
    return __uint_as_float(v);
}

// ---------------------------------------------------------------------------
// Prep kernels
// ---------------------------------------------------------------------------
__global__ void cvt_f32_bf16(const float* __restrict__ src, __bf16* __restrict__ dst, int n) {
    int i = blockIdx.x * blockDim.x + threadIdx.x;
    if (i < n) dst[i] = (__bf16)src[i];
}

// Pack row-major W[K,Ncols] (f32) into WMMA B-operand lane layout (bf16):
// Bp[((kt*NT + nt)*32 + lane)*16 + j] = W[kt*32 + (lane>>4)*16 + j][nt*16 + (lane&15)]
__global__ void pack_b_wmma(const float* __restrict__ W, __bf16* __restrict__ Bp,
                            int K, int Ncols) {
    int idx = blockIdx.x * blockDim.x + threadIdx.x;
    int total = K * Ncols;
    if (idx >= total) return;
    int j    = idx & 15;
    int lane = (idx >> 4) & 31;
    int rest = idx >> 9;
    int NT = Ncols >> 4;
    int nt = rest % NT;
    int kt = rest / NT;
    int k = kt * 32 + (lane >> 4) * 16 + j;
    int n = nt * 16 + (lane & 15);
    Bp[idx] = (__bf16)W[(size_t)k * Ncols + n];
}

__global__ void zero_f32(float* __restrict__ p, size_t n) {
    size_t i = (size_t)blockIdx.x * blockDim.x + threadIdx.x;
    if (i < n) p[i] = 0.0f;
}

// ---------------------------------------------------------------------------
// WMMA GEMM: D[M, Ntiles*16] = A[M,K](bf16, row-major) x Bp(packed bf16)
// One wave32 computes a 16 x (NB*16) strip per K-chunk:
//   - A fragment: one 16x32 bf16 load (2x b128)
//   - all NB B fragments loaded up-front (one clause of 2*NB b128 loads)
//   - NB back-to-back v_wmma_f32_16x16x32_bf16 on independent accumulators
// ---------------------------------------------------------------------------
__global__ void wmma_gemm_bf16(const __bf16* __restrict__ A, const __bf16* __restrict__ Bp,
                               float* __restrict__ D, int Mtiles, int Ntiles, int K) {
    int wave = (blockIdx.x * blockDim.x + threadIdx.x) >> 5;
    int lane = threadIdx.x & 31;
    int Nstrips = Ntiles / NB;
    int total = Mtiles * Nstrips;
    if (wave >= total) return;           // uniform per-wave exit: EXEC stays all-1s for WMMA
    int mt  = wave / Nstrips;
    int nt0 = (wave % Nstrips) * NB;
    int grp = lane >> 4;                 // lane group (0: lanes 0-15, 1: lanes 16-31)
    int row = mt * 16 + (lane & 15);
    const __bf16* arow = A + (size_t)row * K;
    int Ncols = Ntiles << 4;

    v8f c[NB];
#pragma unroll
    for (int nb = 0; nb < NB; ++nb) c[nb] = (v8f){};

    int Kt = K >> 5;
    for (int kt = 0; kt < Kt; ++kt) {
        int k0 = kt << 5;
        if (kt + 1 < Kt) __builtin_prefetch(arow + k0 + 32, 0, 0);
        // A 16x32 bf16 layout: lane grp g -> element j: K = k0+16g+j (matches packed B)
        v16bf a = *(const v16bf*)(arow + k0 + grp * 16);
        // Load all NB B fragments first (consecutive 512-element blocks), then
        // issue the WMMA chain back-to-back on independent accumulators.
        const __bf16* brow = Bp + ((((size_t)kt * Ntiles + nt0) * 32 + lane) << 4);
        v16bf b[NB];
#pragma unroll
        for (int nb = 0; nb < NB; ++nb)
            b[nb] = *(const v16bf*)(brow + ((size_t)nb * 32 << 4));
#pragma unroll
        for (int nb = 0; nb < NB; ++nb)
            c[nb] = __builtin_amdgcn_wmma_f32_16x16x32_bf16(false, a, false, b[nb],
                                                            (short)0, c[nb], false, false);
    }
    // C/D layout: VGPR r -> (M = grp*8 + r, N = lane&15)
    int colb = lane & 15;
    float* dbase = D + (size_t)(mt * 16 + grp * 8) * Ncols;
#pragma unroll
    for (int nb = 0; nb < NB; ++nb) {
        float* dp = dbase + (nt0 + nb) * 16 + colb;
#pragma unroll
        for (int r = 0; r < 8; ++r) dp[(size_t)r * Ncols] = c[nb][r];
    }
}

// ---------------------------------------------------------------------------
// GAT edge machinery (PyG semantics: gather -> leaky_relu -> segment softmax
// over dst -> scatter-add). Self-loops are appended logically: e >= E => (e-E).
// ---------------------------------------------------------------------------
__global__ void node_dots(const float* __restrict__ hmat, const float* __restrict__ asrc,
                          const float* __restrict__ adst, float* __restrict__ es,
                          float* __restrict__ ed, int N, int H, int C) {
    int idx = blockIdx.x * blockDim.x + threadIdx.x;
    if (idx >= N * H) return;
    int n = idx / H, h = idx % H;
    const float* hp = hmat + (size_t)n * H * C + (size_t)h * C;
    float s = 0.f, d = 0.f;
    for (int c = 0; c < C; ++c) {
        float v = hp[c];
        s = fmaf(v, asrc[h * C + c], s);
        d = fmaf(v, adst[h * C + c], d);
    }
    es[idx] = s; ed[idx] = d;
}

__global__ void edge_score_max(const int* __restrict__ ei, int E, int H,
                               const float* __restrict__ es, const float* __restrict__ ed,
                               float* __restrict__ score, unsigned* __restrict__ mU,
                               long long total) {
    long long idx = (long long)blockIdx.x * blockDim.x + threadIdx.x;
    if (idx >= total) return;
    int e = (int)(idx / H), h = (int)(idx % H);
    int s = (e < E) ? ei[e] : (e - E);
    int d = (e < E) ? ei[E + e] : (e - E);
    float v = es[(size_t)s * H + h] + ed[(size_t)d * H + h];
    v = (v > 0.f) ? v : 0.2f * v;                     // leaky_relu(0.2)
    score[idx] = v;
    atomicMax(&mU[(size_t)d * H + h], f2ord(v));      // segment max (ordered-uint)
}

__global__ void edge_exp_sum(const int* __restrict__ ei, int E, int H,
                             float* __restrict__ score, const unsigned* __restrict__ mU,
                             float* __restrict__ den, long long total) {
    long long idx = (long long)blockIdx.x * blockDim.x + threadIdx.x;
    if (idx >= total) return;
    int e = (int)(idx / H), h = (int)(idx % H);
    int d = (e < E) ? ei[E + e] : (e - E);
    float ex = expf(score[idx] - ord2f(mU[(size_t)d * H + h]));
    score[idx] = ex;                                   // reuse buffer: now holds exp()
    atomicAdd(&den[(size_t)d * H + h], ex);
}

__global__ void edge_scatter(const int* __restrict__ ei, int E, int H, int C,
                             const float* __restrict__ ex, const float* __restrict__ den,
                             const float* __restrict__ hmat, float* __restrict__ out,
                             long long total) {
    long long idx = (long long)blockIdx.x * blockDim.x + threadIdx.x;
    if (idx >= total) return;
    int HC = H * C;
    int e = (int)(idx / HC), hc = (int)(idx % HC), h = hc / C;
    int s = (e < E) ? ei[e] : (e - E);
    int d = (e < E) ? ei[E + e] : (e - E);
    float a = ex[(size_t)e * H + h] / (den[(size_t)d * H + h] + 1e-16f);
    atomicAdd(&out[(size_t)d * HC + hc], hmat[(size_t)s * HC + hc] * a);
}

__global__ void alpha_write(const int* __restrict__ ei, int E,
                            const float* __restrict__ ex, const float* __restrict__ den,
                            float* __restrict__ alpha, int Etot) {
    int e = blockIdx.x * blockDim.x + threadIdx.x;
    if (e >= Etot) return;
    int d = (e < E) ? ei[E + e] : (e - E);
    alpha[e] = ex[e] / (den[d] + 1e-16f);
}

// ---------------------------------------------------------------------------
// out + bias -> (optional ELU) -> LayerNorm(C) ; per-node block, C = 64/256
// ---------------------------------------------------------------------------
__global__ void bias_act_ln(const float* __restrict__ acc, const float* __restrict__ bias,
                            const float* __restrict__ g, const float* __restrict__ b,
                            float* __restrict__ outF, __bf16* __restrict__ outB,
                            int C, int doElu) {
    extern __shared__ float sh[];
    int n = blockIdx.x, t = threadIdx.x;
    float v = acc[(size_t)n * C + t] + bias[t];
    if (doElu) v = (v > 0.f) ? v : (expf(v) - 1.f);
    sh[t] = v; __syncthreads();
    for (int s = C >> 1; s > 0; s >>= 1) { if (t < s) sh[t] += sh[t + s]; __syncthreads(); }
    float mu = sh[0] / C; __syncthreads();
    float dv = v - mu;
    sh[t] = dv * dv; __syncthreads();
    for (int s = C >> 1; s > 0; s >>= 1) { if (t < s) sh[t] += sh[t + s]; __syncthreads(); }
    float y = dv * rsqrtf(sh[0] / C + 1e-5f) * g[t] + b[t];
    if (outF) outF[(size_t)n * C + t] = y;
    if (outB) outB[(size_t)n * C + t] = (__bf16)y;
}

// ---------------------------------------------------------------------------
// Dueling head (single block, tiny)
// ---------------------------------------------------------------------------
static __device__ void ln_shared(float* arr, int len, const float* g, const float* b,
                                 float* red) {
    __syncthreads();
    if (threadIdx.x == 0) {
        float s = 0.f; for (int i = 0; i < len; ++i) s += arr[i];
        float mu = s / len;
        float v = 0.f; for (int i = 0; i < len; ++i) { float d = arr[i] - mu; v += d * d; }
        red[0] = mu; red[1] = rsqrtf(v / len + 1e-5f);
    }
    __syncthreads();
    if ((int)threadIdx.x < len)
        arr[threadIdx.x] = (arr[threadIdx.x] - red[0]) * red[1] * g[threadIdx.x] + b[threadIdx.x];
    __syncthreads();
}

__global__ void head_kernel(const float* __restrict__ h2, const int* __restrict__ activeIdx,
                            const int* __restrict__ stepPtr, const float* __restrict__ semb,
                            const float* __restrict__ fc0w, const float* __restrict__ fc0b,
                            const float* __restrict__ fc1w, const float* __restrict__ fc1b,
                            const float* __restrict__ fc2w, const float* __restrict__ fc2b,
                            const float* __restrict__ fc3w, const float* __restrict__ fc3b,
                            const float* __restrict__ valw, const float* __restrict__ valb,
                            const float* __restrict__ advw, const float* __restrict__ advb,
                            const float* __restrict__ lnhg, const float* __restrict__ lnhb,
                            const float* __restrict__ lnfg, const float* __restrict__ lnfb,
                            float* __restrict__ logits) {
    __shared__ float sent[64], st[64], anf[128], anf2[64], adv[32], red[2];
    int t = threadIdx.x;
    // sent = LN(relu(semb @ fc1 + b))
    if (t < 64) {
        float acc = fc1b[t];
        for (int k = 0; k < SENT; ++k) acc = fmaf(semb[k], fc1w[k * 64 + t], acc);
        sent[t] = fmaxf(acc, 0.f);
    }
    ln_shared(sent, 64, lnhg, lnhb, red);
    // steps = LN(relu((step+1)*0.01 * fc0 + b))
    if (t < 64) {
        float s0 = ((float)stepPtr[0] + 1.f) * 0.01f;
        st[t] = fmaxf(fmaf(s0, fc0w[t], fc0b[t]), 0.f);
    }
    ln_shared(st, 64, lnhg, lnhb, red);
    if (t < 64) sent[t] += st[t];
    __syncthreads();
    // sent = LN(relu(sent @ fc2 + b))
    if (t < 64) {
        float acc = fc2b[t];
        for (int k = 0; k < 64; ++k) acc = fmaf(sent[k], fc2w[k * 64 + t], acc);
        st[t] = fmaxf(acc, 0.f);
    }
    ln_shared(st, 64, lnhg, lnhb, red);
    // anf = LN_f(concat(h2[active], sent))
    if (t < 64) {
        int an = activeIdx[0];
        anf[t]      = h2[(size_t)an * 64 + t];
        anf[64 + t] = st[t];
    }
    ln_shared(anf, 128, lnfg, lnfb, red);
    // anf = LN(relu(anf @ fc3 + b))
    if (t < 64) {
        float acc = fc3b[t];
        for (int k = 0; k < 128; ++k) acc = fmaf(anf[k], fc3w[k * 64 + t], acc);
        anf2[t] = fmaxf(acc, 0.f);
    }
    ln_shared(anf2, 64, lnhg, lnhb, red);
    // dueling: tanh(val + adv - mean(adv))
    if (t < ACT) {
        float acc = advb[t];
        for (int k = 0; k < 64; ++k) acc = fmaf(anf2[k], advw[k * ACT + t], acc);
        adv[t] = acc;
    }
    __syncthreads();
    if (t == 0) {
        float val = valb[0];
        for (int k = 0; k < 64; ++k) val = fmaf(anf2[k], valw[k], val);
        float m = 0.f; for (int k = 0; k < ACT; ++k) m += adv[k];
        red[0] = val - m / (float)ACT;
    }
    __syncthreads();
    if (t < ACT) logits[t] = tanhf(red[0] + adv[t]);
}

// ---------------------------------------------------------------------------
// Launch
// ---------------------------------------------------------------------------
static inline unsigned cdiv(long long a, long long b) { return (unsigned)((a + b - 1) / b); }

extern "C" void kernel_launch(void* const* d_in, const int* in_sizes, int n_in,
                              void* d_out, int out_size, void* d_ws, size_t ws_size,
                              hipStream_t stream) {
    const float* x     = (const float*)d_in[0];
    const int*   ei    = (const int*)d_in[1];
    const float* semb  = (const float*)d_in[2];
    const int*   activ = (const int*)d_in[3];
    const int*   stepP = (const int*)d_in[4];
    const float* W1    = (const float*)d_in[5];
    const float* as1   = (const float*)d_in[6];
    const float* ad1   = (const float*)d_in[7];
    const float* b1    = (const float*)d_in[8];
    const float* W2    = (const float*)d_in[9];
    const float* as2   = (const float*)d_in[10];
    const float* ad2   = (const float*)d_in[11];
    const float* b2    = (const float*)d_in[12];
    const float* fc0w  = (const float*)d_in[13];
    const float* fc0b  = (const float*)d_in[14];
    const float* fc1w  = (const float*)d_in[15];
    const float* fc1b  = (const float*)d_in[16];
    const float* fc2w  = (const float*)d_in[17];
    const float* fc2b  = (const float*)d_in[18];
    const float* fc3w  = (const float*)d_in[19];
    const float* fc3b  = (const float*)d_in[20];
    const float* valw  = (const float*)d_in[21];
    const float* valb  = (const float*)d_in[22];
    const float* advw  = (const float*)d_in[23];
    const float* advb  = (const float*)d_in[24];
    const float* lnhg  = (const float*)d_in[25];
    const float* lnhb  = (const float*)d_in[26];
    const float* lnfg  = (const float*)d_in[27];
    const float* lnfb  = (const float*)d_in[28];
    const float* lnag  = (const float*)d_in[29];
    const float* lnab  = (const float*)d_in[30];

    const int N = in_sizes[0] / F_IN;        // 50000
    const int E = in_sizes[1] / 2;           // 600000
    const int Etot = E + N;                  // with self-loops
    const int HC1 = HEADS * HIDC;            // 256

    // ---- workspace carve (256-byte aligned) ----
    char* w = (char*)d_ws;
    auto carve = [&](size_t bytes) -> void* {
        void* p = (void*)w;
        w += (bytes + 255) & ~(size_t)255;
        return p;
    };
    __bf16* xbf   = (__bf16*)carve((size_t)N * F_IN * 2);
    __bf16* w1p   = (__bf16*)carve((size_t)F_IN * HC1 * 2);
    __bf16* w2p   = (__bf16*)carve((size_t)HC1 * HIDC * 2);
    float*  h1pre = (float*)carve((size_t)N * HC1 * 4);
    __bf16* h1bf  = (__bf16*)carve((size_t)N * HC1 * 2);
    float*  h2pre = (float*)carve((size_t)N * HIDC * 4);
    float*  es1   = (float*)carve((size_t)N * HEADS * 4);
    float*  ed1   = (float*)carve((size_t)N * HEADS * 4);
    float*  es2   = (float*)carve((size_t)N * 4);
    float*  ed2   = (float*)carve((size_t)N * 4);
    float*  sc1   = (float*)carve((size_t)Etot * HEADS * 4);  // score -> exp
    float*  sc2   = (float*)carve((size_t)Etot * 4);
    char* zbeg = w;                                            // zero-init region
    unsigned* m1u = (unsigned*)carve((size_t)N * HEADS * 4);
    float*    den1 = (float*)carve((size_t)N * HEADS * 4);
    float*    out1 = (float*)carve((size_t)N * HC1 * 4);
    unsigned* m2u = (unsigned*)carve((size_t)N * 4);
    float*    den2 = (float*)carve((size_t)N * 4);
    float*    out2 = (float*)carve((size_t)N * HIDC * 4);
    size_t zelems = (size_t)(w - zbeg) / 4;

    float* outLogits = (float*)d_out;
    float* outH2     = outLogits + 32;
    float* outAlpha  = outH2 + (size_t)N * HIDC;

    const int T = 256;

    // ---- prep ----
    cvt_f32_bf16<<<cdiv((long long)N * F_IN, T), T, 0, stream>>>(x, xbf, N * F_IN);
    pack_b_wmma<<<cdiv(F_IN * HC1, T), T, 0, stream>>>(W1, w1p, F_IN, HC1);
    pack_b_wmma<<<cdiv(HC1 * HIDC, T), T, 0, stream>>>(W2, w2p, HC1, HIDC);
    zero_f32<<<cdiv((long long)zelems, T), T, 0, stream>>>((float*)zbeg, zelems);

    // ---- GAT layer 1 ----
    {
        int Mt = N / 16, Nt = HC1 / 16;                    // 3125 x 16 tiles
        long long waves = (long long)Mt * (Nt / NB);       // 16x64 strip per wave
        wmma_gemm_bf16<<<cdiv(waves * 32, T), T, 0, stream>>>(xbf, w1p, h1pre, Mt, Nt, F_IN);
    }
    node_dots<<<cdiv((long long)N * HEADS, T), T, 0, stream>>>(h1pre, as1, ad1, es1, ed1, N, HEADS, HIDC);
    {
        long long tot = (long long)Etot * HEADS;
        edge_score_max<<<cdiv(tot, T), T, 0, stream>>>(ei, E, HEADS, es1, ed1, sc1, m1u, tot);
        edge_exp_sum<<<cdiv(tot, T), T, 0, stream>>>(ei, E, HEADS, sc1, m1u, den1, tot);
        long long tot2 = (long long)Etot * HC1;
        edge_scatter<<<cdiv(tot2, T), T, 0, stream>>>(ei, E, HEADS, HIDC, sc1, den1, h1pre, out1, tot2);
    }
    bias_act_ln<<<N, HC1, HC1 * 4, stream>>>(out1, b1, lnag, lnab, nullptr, h1bf, HC1, 1);

    // ---- GAT layer 2 ----
    {
        int Mt = N / 16, Nt = HIDC / 16;                   // 3125 x 4 tiles
        long long waves = (long long)Mt * (Nt / NB);       // 16x64 strip per wave
        wmma_gemm_bf16<<<cdiv(waves * 32, T), T, 0, stream>>>(h1bf, w2p, h2pre, Mt, Nt, HC1);
    }
    node_dots<<<cdiv((long long)N, T), T, 0, stream>>>(h2pre, as2, ad2, es2, ed2, N, 1, HIDC);
    {
        long long tot = (long long)Etot;
        edge_score_max<<<cdiv(tot, T), T, 0, stream>>>(ei, E, 1, es2, ed2, sc2, m2u, tot);
        edge_exp_sum<<<cdiv(tot, T), T, 0, stream>>>(ei, E, 1, sc2, m2u, den2, tot);
        alpha_write<<<cdiv(tot, T), T, 0, stream>>>(ei, E, sc2, den2, outAlpha, Etot);
        long long tot2 = (long long)Etot * HIDC;
        edge_scatter<<<cdiv(tot2, T), T, 0, stream>>>(ei, E, 1, HIDC, sc2, den2, h2pre, out2, tot2);
    }
    bias_act_ln<<<N, HIDC, HIDC * 4, stream>>>(out2, b2, lnhg, lnhb, outH2, nullptr, HIDC, 0);

    // ---- dueling head ----
    head_kernel<<<1, 128, 0, stream>>>(outH2, activ, stepP, semb,
                                       fc0w, fc0b, fc1w, fc1b, fc2w, fc2b, fc3w, fc3b,
                                       valw, valb, advw, advb,
                                       lnhg, lnhb, lnfg, lnfb, outLogits);
}
